// SEAL_45475113730061
// MI455X (gfx1250) — matrix operand order repaired and checked by reading, not compiled, not run on hardware
//
#include <hip/hip_runtime.h>
#include <hip/hip_bf16.h>
#include <math.h>
#include <stdint.h>

typedef _Float16 f16;
typedef unsigned int u32;
typedef __attribute__((ext_vector_type(16))) _Float16     v16h;
typedef __attribute__((ext_vector_type(8)))  float        v8f;
typedef __attribute__((ext_vector_type(4)))  unsigned int v4u;
typedef __attribute__((ext_vector_type(8)))  int          v8i;
typedef __attribute__((ext_vector_type(4)))  int          v4i;

#define G_   1000
#define N_   256
#define E_   1024
#define F_   128
#define D1_  128
#define D2_  64
#define K1_  32
#define K2_  8
#define ME_  16000
#define MD_  128
#define LB_  10

// ---------------- TDM: async 2D f32 tile load (global -> LDS), ISA §8 D# layout ---------
// group0: count=1 | lds_addr | global_addr[31:0] | global_addr[56:32]|type=2
// group1: data_size=4B | tensor_dim0/1 | tile_dim0/1 (tile_dim2=0) | dim0 stride
// groups 2/3 (and the trailing int32x8 group on this toolchain) unused -> zeros (<=2D tensor)
__device__ inline void tdm_load_2d_f32(u32 lds_addr, const float* gsrc,
                                       u32 tensor_w, u32 tensor_h,
                                       u32 tile_w, u32 tile_h, u32 stride_elems) {
  unsigned long long ga = (unsigned long long)(uintptr_t)gsrc;
  v4u g0;
  g0[0] = 1u;                                             // count=1, user descriptor
  g0[1] = lds_addr;                                       // LDS byte address
  g0[2] = (u32)(ga & 0xFFFFFFFFull);                      // global_addr lo
  g0[3] = (u32)((ga >> 32) & 0x1FFFFFFull) | (2u << 30);  // global_addr hi | type=2
  v8i g1;
  g1[0] = (int)(2u << 16);                                // data_size = 4 bytes
  g1[1] = (int)((tensor_w & 0xFFFFu) << 16);              // tensor_dim0[15:0]
  g1[2] = (int)(((tensor_w >> 16) & 0xFFFFu) | ((tensor_h & 0xFFFFu) << 16));
  g1[3] = (int)(((tensor_h >> 16) & 0xFFFFu) | ((tile_w & 0xFFFFu) << 16));   // | tile_dim0
  g1[4] = (int)(tile_h & 0xFFFFu);                        // tile_dim1 (tile_dim2=0)
  g1[5] = (int)stride_elems;                              // tensor_dim0_stride lo
  g1[6] = 0;
  g1[7] = 0;
  v4i z4 = {0, 0, 0, 0};
  v8i z8 = {0, 0, 0, 0, 0, 0, 0, 0};
  __builtin_amdgcn_tensor_load_to_lds(g0, g1, z4, z4, z8, 0);
}

// ---------------- WMMA fragment helpers (wave32, gfx1250 layouts) ----------------

__device__ inline v16h load_a_frag(const f16* sA, int lda, int m0, int k0, int lane) {
  v16h a;
  int row = m0 + (lane & 15);
  int kb  = k0 + ((lane >> 4) << 3);      // +8 for lanes 16..31
#pragma unroll
  for (int i = 0; i < 16; ++i) {
    int v = i >> 1, pos = i & 1;
    int k = kb + ((v >> 2) << 4) + ((v & 3) << 1) + pos;
    a[i] = sA[row * lda + k];
  }
  return a;
}

__device__ inline v16h load_b_frag(const f16* sB, int ldb, int k0, int n0, int lane) {
  v16h b;
  int col = n0 + (lane & 15);
  int kb  = k0 + ((lane >> 4) << 3);      // +8 for lanes 16..31
#pragma unroll
  for (int i = 0; i < 16; ++i)
    b[i] = sB[(kb + (i & 7) + ((i >> 3) << 4)) * ldb + col];
  return b;
}

// C = A(MxK,f16,LDS) @ B(KxN,f16,LDS), f16 output to LDS (ld=N). 8 waves/block.
// Tile loop is scalar (readfirstlane) so EXEC stays all-ones around every WMMA.
template<int M, int K, int Nn>
__device__ inline void lds_gemm_f16out(const f16* sA, const f16* sB, f16* sOut) {
  int tid = threadIdx.x, lane = tid & 31;
  int wave = __builtin_amdgcn_readfirstlane(tid >> 5);
  const int NT = Nn / 16, TILES = (M / 16) * NT;
  for (int t = wave; t < TILES; t += 8) {
    int m0 = (t / NT) * 16, n0 = (t % NT) * 16;
    v8f acc = {0.f,0.f,0.f,0.f,0.f,0.f,0.f,0.f};
#pragma unroll
    for (int k0 = 0; k0 < K; k0 += 32) {
      v16h a = load_a_frag(sA, K, m0, k0, lane);
      v16h b = load_b_frag(sB, Nn, k0, n0, lane);
      acc = __builtin_amdgcn_wmma_f32_16x16x32_f16(false, a, false, b,
                                                   (short)0, acc, false, false);
    }
    int col = n0 + (lane & 15);
    int rb  = m0 + ((lane >> 4) << 3);
#pragma unroll
    for (int r = 0; r < 8; ++r) sOut[(rb + r) * Nn + col] = (f16)acc[r];
  }
}

// Same but f32 output to LDS.
template<int M, int K, int Nn>
__device__ inline void lds_gemm_f32out(const f16* sA, const f16* sB, float* sOut) {
  int tid = threadIdx.x, lane = tid & 31;
  int wave = __builtin_amdgcn_readfirstlane(tid >> 5);
  const int NT = Nn / 16, TILES = (M / 16) * NT;
  for (int t = wave; t < TILES; t += 8) {
    int m0 = (t / NT) * 16, n0 = (t % NT) * 16;
    v8f acc = {0.f,0.f,0.f,0.f,0.f,0.f,0.f,0.f};
#pragma unroll
    for (int k0 = 0; k0 < K; k0 += 32) {
      v16h a = load_a_frag(sA, K, m0, k0, lane);
      v16h b = load_b_frag(sB, Nn, k0, n0, lane);
      acc = __builtin_amdgcn_wmma_f32_16x16x32_f16(false, a, false, b,
                                                   (short)0, acc, false, false);
    }
    int col = n0 + (lane & 15);
    int rb  = m0 + ((lane >> 4) << 3);
#pragma unroll
    for (int r = 0; r < 8; ++r) sOut[(rb + r) * Nn + col] = acc[r];
  }
}

// Edge scatter: sC[d][:] += dinv[s]*dinv[d] * sXW[s][:]  (LDS atomics)
template<int DD>
__device__ inline void lds_aggregate(const int* src, const int* dst,
                                     const f16* sXW, float* sC, const float* sDinv) {
  int tid = threadIdx.x, lane = tid & 31;
  int wave = __builtin_amdgcn_readfirstlane(tid >> 5);
  for (int e = wave; e < E_; e += 8) {
    int s = src[e], d = dst[e];
    float nrm = sDinv[s] * sDinv[d];
#pragma unroll
    for (int f = lane; f < DD; f += 32)
      atomicAdd(&sC[d * DD + f], nrm * (float)sXW[s * DD + f]);
  }
}

// ---------------- Stage 1: one workgroup per graph, fully LDS-resident ----------------

__global__ __launch_bounds__(256) void seal_graph_kernel(
    const float* __restrict__ gx, const int* __restrict__ gedges,
    const float* __restrict__ W1, const float* __restrict__ b1,
    const float* __restrict__ W2, const float* __restrict__ b2,
    const float* __restrict__ Wf1, const float* __restrict__ bf1,
    const float* __restrict__ Wf2, const float* __restrict__ bf2,
    float* __restrict__ embs, float* __restrict__ pens)
{
  __shared__ f16   sA[N_ * F_];      // 64 KB: f16 activations (WMMA A)
  __shared__ f16   sXW[N_ * F_];     // 64 KB: GEMM outputs = per-node messages
  __shared__ f16   sB[F_ * D1_];     // 32 KB: weights (WMMA B)
  __shared__ float sC[N_ * D1_];     // 128 KB: TDM staging, then f32 aggregation accum
  __shared__ float sDinv[N_];
  __shared__ int   sDeg[N_];
  __shared__ float sAtt[N_ * K2_];   // 8 KB
  __shared__ float sWf2[K1_ * K2_];
  __shared__ float sP[K2_ * K2_];
  __shared__ float sRed[K2_];

  const int g = blockIdx.x, tid = threadIdx.x;
  const int lane = tid & 31;
  const int wave = __builtin_amdgcn_readfirstlane(tid >> 5);
  const float* x  = gx + (size_t)g * N_ * F_;
  const int* src  = gedges + (size_t)g * 2 * E_;
  const int* dst  = src + E_;

  __builtin_prefetch(W1 + tid * 64, 0, 3);   // global_prefetch_b8: warm W1 (64KB)

  if (tid < N_) sDeg[tid] = 1;
  __syncthreads();

  // TDM: async-load x[256,128] f32 into sC staging (one descriptor, wave 0 issues)
  if (wave == 0)
    tdm_load_2d_f32((u32)(uintptr_t)(void*)sC, x, F_, N_, F_, N_, F_);

  // meanwhile: degrees + weights (overlap with the TDM transfer)
  for (int e = tid; e < E_; e += 256) atomicAdd(&sDeg[dst[e]], 1);
  for (int i = tid; i < F_ * D1_; i += 256) sB[i] = (f16)W1[i];
  if (wave == 0) __builtin_amdgcn_s_wait_tensorcnt(0);
  __syncthreads();

  if (tid < N_) sDinv[tid] = rsqrtf((float)sDeg[tid]);
  for (int i = tid; i < N_ * F_; i += 256) sA[i] = (f16)sC[i];   // x -> f16
  __syncthreads();

  // ---- layer 1: xw1 = x @ W1 -> sXW (256x128)
  lds_gemm_f16out<N_, F_, D1_>(sA, sB, sXW);
  __syncthreads();
  for (int i = tid; i < N_ * D1_; i += 256) {
    int n = i / D1_, f = i - n * D1_;
    float di = sDinv[n];
    sC[i] = b1[f] + di * di * (float)sXW[i];
  }
  __syncthreads();
  lds_aggregate<D1_>(src, dst, sXW, sC, sDinv);
  __syncthreads();

  // h1 = relu(sC) -> sA f16 ; load W2
  for (int i = tid; i < N_ * D1_; i += 256) sA[i] = (f16)fmaxf(sC[i], 0.f);
  for (int i = tid; i < D1_ * D2_; i += 256) sB[i] = (f16)W2[i];
  __syncthreads();

  // ---- layer 2: xw2 = h1 @ W2 -> sXW (256x64)
  lds_gemm_f16out<N_, D1_, D2_>(sA, sB, sXW);
  __syncthreads();
  for (int i = tid; i < N_ * D2_; i += 256) {
    int n = i / D2_, f = i - n * D2_;
    float di = sDinv[n];
    sC[i] = b2[f] + di * di * (float)sXW[i];
  }
  __syncthreads();
  lds_aggregate<D2_>(src, dst, sXW, sC, sDinv);
  __syncthreads();
  // h2 (f32) in sC[0 .. 256*64)

  // ---- attention: a = tanh(h2 @ Wf1 + bf1)
  for (int i = tid; i < N_ * D2_; i += 256) sA[i] = (f16)sC[i];
  for (int i = tid; i < D2_ * K1_; i += 256) sB[i] = (f16)Wf1[i];
  __syncthreads();
  float* sAf = sC + N_ * D2_;                       // 256x32 f32 region after h2
  lds_gemm_f32out<N_, D2_, K1_>(sA, sB, sAf);
  __syncthreads();
  for (int i = tid; i < N_ * K1_; i += 256) sAf[i] = tanhf(sAf[i] + bf1[i & (K1_ - 1)]);
  if (tid < K1_ * K2_) sWf2[tid] = Wf2[tid];
  __syncthreads();

  // att logits: z[n,k] = a[n,:] @ Wf2[:,k] + bf2[k]   (one node per thread)
  {
    int n = tid;
    float acc[K2_];
#pragma unroll
    for (int k = 0; k < K2_; ++k) acc[k] = bf2[k];
#pragma unroll
    for (int j = 0; j < K1_; ++j) {
      float av = sAf[n * K1_ + j];
#pragma unroll
      for (int k = 0; k < K2_; ++k) acc[k] += av * sWf2[j * K2_ + k];
    }
#pragma unroll
    for (int k = 0; k < K2_; ++k) sAtt[n * K2_ + k] = acc[k];
  }
  __syncthreads();

  // softmax over nodes (axis=0): wave w owns column w (8 waves, 8 columns)
  {
    int k = wave;
    float m = -1e30f;
    for (int n = lane; n < N_; n += 32) m = fmaxf(m, sAtt[n * K2_ + k]);
    for (int off = 16; off; off >>= 1) m = fmaxf(m, __shfl_xor(m, off, 32));
    float s = 0.f;
    for (int n = lane; n < N_; n += 32) s += expf(sAtt[n * K2_ + k] - m);
    for (int off = 16; off; off >>= 1) s += __shfl_xor(s, off, 32);
    float inv = 1.f / s;
    for (int n = lane; n < N_; n += 32)
      sAtt[n * K2_ + k] = expf(sAtt[n * K2_ + k] - m) * inv;
  }
  __syncthreads();

  // emb[k*64+d] = sum_n att[n,k] * h2[n,d]
  for (int o = tid; o < K2_ * D2_; o += 256) {
    int k = o >> 6, d = o & 63;
    float s = 0.f;
    for (int n = 0; n < N_; ++n) s += sAtt[n * K2_ + k] * sC[n * D2_ + d];
    embs[(size_t)g * (K2_ * D2_) + o] = s;
  }

  // penalty: P = att.T @ att - I ; pen = sum_i ||P[i,:]||_2
  if (tid < K2_ * K2_) {
    int i = tid >> 3, j = tid & 7;
    float s = 0.f;
    for (int n = 0; n < N_; ++n) s += sAtt[n * K2_ + i] * sAtt[n * K2_ + j];
    if (i == j) s -= 1.f;
    sP[tid] = s;
  }
  __syncthreads();
  if (tid < K2_) {
    float s = 0.f;
#pragma unroll
    for (int j = 0; j < K2_; ++j) s += sP[tid * K2_ + j] * sP[tid * K2_ + j];
    sRed[tid] = sqrtf(s);
  }
  __syncthreads();
  if (tid == 0) {
    float s = 0.f;
#pragma unroll
    for (int i = 0; i < K2_; ++i) s += sRed[i];
    pens[g] = s;
  }
}

// ---------------- Stage 2: macro GCN ----------------

__global__ __launch_bounds__(256) void seal_macro_deg_init(int* deg) {
  int i = blockIdx.x * blockDim.x + threadIdx.x;
  if (i < G_) deg[i] = 1;
}
__global__ __launch_bounds__(256) void seal_macro_deg_count(const int* mdst, int* deg) {
  int e = blockIdx.x * blockDim.x + threadIdx.x;
  if (e < ME_) atomicAdd(&deg[mdst[e]], 1);
}
__global__ __launch_bounds__(256) void seal_macro_dinv(const int* deg, float* dinv) {
  int i = blockIdx.x * blockDim.x + threadIdx.x;
  if (i < G_) dinv[i] = rsqrtf((float)deg[i]);
}

// xw = embs[1000,512] @ Wm1[512,128] — WMMA, 64 rows/block, TDM-staged A chunks
__global__ __launch_bounds__(256) void seal_macro_gemm(
    const float* __restrict__ embs, const float* __restrict__ Wm1, float* __restrict__ xw) {
  __shared__ f16   sA[64 * 128];     // 16 KB
  __shared__ f16   sB[128 * 128];    // 32 KB
  __shared__ float stgA[64 * 128];   // 32 KB TDM staging
  int tid = threadIdx.x, lane = tid & 31;
  int wave = __builtin_amdgcn_readfirstlane(tid >> 5);
  int row0 = blockIdx.x * 64;
  u32 rows = (u32)((G_ - row0 < 64) ? (G_ - row0) : 64);   // ragged tail -> TDM OOB zeros
  v8f acc[4];
#pragma unroll
  for (int t = 0; t < 4; ++t) acc[t] = (v8f){0.f,0.f,0.f,0.f,0.f,0.f,0.f,0.f};

  for (int kc = 0; kc < 512; kc += 128) {
    // TDM: 64x128 f32 tile out of the 512-wide embs tensor (strided)
    if (wave == 0)
      tdm_load_2d_f32((u32)(uintptr_t)(void*)stgA, embs + (size_t)row0 * 512 + kc,
                      128u, rows, 128u, 64u, 512u);
    // B chunk (contiguous) via regular loads, overlapped with the DMA
    for (int i = tid; i < 128 * 128; i += 256) {
      int r = i >> 7, c = i & 127;
      sB[i] = (f16)Wm1[(size_t)(kc + r) * 128 + c];
    }
    if (wave == 0) __builtin_amdgcn_s_wait_tensorcnt(0);
    __syncthreads();
    for (int i = tid; i < 64 * 128; i += 256) sA[i] = (f16)stgA[i];
    __syncthreads();
#pragma unroll
    for (int tl = 0; tl < 4; ++tl) {
      int m0 = tl * 16, n0 = wave * 16;
#pragma unroll
      for (int k0 = 0; k0 < 128; k0 += 32) {
        v16h a = load_a_frag(sA, 128, m0, k0, lane);
        v16h b = load_b_frag(sB, 128, k0, n0, lane);
        acc[tl] = __builtin_amdgcn_wmma_f32_16x16x32_f16(false, a, false, b,
                                                         (short)0, acc[tl], false, false);
      }
    }
    __syncthreads();
  }
#pragma unroll
  for (int tl = 0; tl < 4; ++tl) {
    int m0 = tl * 16, n0 = wave * 16;
    int col = n0 + (lane & 15);
    int rb  = m0 + ((lane >> 4) << 3);
#pragma unroll
    for (int r = 0; r < 8; ++r) {
      int row = row0 + rb + r;
      if (row < G_) xw[(size_t)row * 128 + col] = acc[tl][r];
    }
  }
}

__global__ void seal_agg1_init(float* h, const float* xw, const float* dinv, const float* bm1) {
  int g = blockIdx.x, f = threadIdx.x;
  float di = dinv[g];
  h[g * MD_ + f] = bm1[f] + di * di * xw[g * MD_ + f];
}
__global__ void seal_agg1_edges(float* h, const float* xw, const float* dinv,
                                const int* msrc, const int* mdst) {
  int e = blockIdx.x, f = threadIdx.x;
  int s = msrc[e], d = mdst[e];
  float nrm = dinv[s] * dinv[d];
  atomicAdd(&h[d * MD_ + f], nrm * xw[s * MD_ + f]);
}

// xwl = relu(h) @ Wm2[128,10]
__global__ void seal_small_gemm(const float* h, const float* Wm2, float* xwl) {
  __shared__ float sh[MD_];
  int g = blockIdx.x, tid = threadIdx.x;
  sh[tid] = fmaxf(h[g * MD_ + tid], 0.f);
  __syncthreads();
  if (tid < LB_) {
    float s = 0.f;
    for (int f = 0; f < MD_; ++f) s += sh[f] * Wm2[f * LB_ + tid];
    xwl[g * LB_ + tid] = s;
  }
}

__global__ void seal_agg2_init(float* logits, const float* xwl, const float* dinv, const float* bm2) {
  int g = blockIdx.x, c = threadIdx.x;
  if (c < LB_) {
    float di = dinv[g];
    logits[g * LB_ + c] = bm2[c] + di * di * xwl[g * LB_ + c];
  }
}
__global__ void seal_agg2_edges(float* logits, const float* xwl, const float* dinv,
                                const int* msrc, const int* mdst) {
  int e = blockIdx.x, c = threadIdx.x;
  if (c < LB_) {
    int s = msrc[e], d = mdst[e];
    atomicAdd(&logits[d * LB_ + c], dinv[s] * dinv[d] * xwl[s * LB_ + c]);
  }
}

__global__ void seal_logsoftmax(const float* logits, float* out) {
  int g = blockIdx.x, lane = threadIdx.x;
  float v = (lane < LB_) ? logits[g * LB_ + lane] : -1e30f;
  float m = v;
  for (int off = 16; off; off >>= 1) m = fmaxf(m, __shfl_xor(m, off, 32));
  float e = (lane < LB_) ? expf(v - m) : 0.f;
  float s = e;
  for (int off = 16; off; off >>= 1) s += __shfl_xor(s, off, 32);
  if (lane < LB_) out[g * LB_ + lane] = v - m - logf(s);
}

__global__ __launch_bounds__(256) void seal_pen_mean(const float* pens, float* out) {
  __shared__ float sr[256];
  int tid = threadIdx.x;
  float s = 0.f;
  for (int i = tid; i < G_; i += 256) s += pens[i];
  sr[tid] = s;
  __syncthreads();
  for (int off = 128; off; off >>= 1) {
    if (tid < off) sr[tid] += sr[tid + off];
    __syncthreads();
  }
  if (tid == 0) out[0] = sr[0] * (1.0f / (float)G_);
}

// ---------------- launch ----------------

extern "C" void kernel_launch(void* const* d_in, const int* in_sizes, int n_in,
                              void* d_out, int out_size, void* d_ws, size_t ws_size,
                              hipStream_t stream) {
  const float* gx     = (const float*)d_in[0];
  const int*   gedges = (const int*)d_in[1];
  const int*   medges = (const int*)d_in[2];
  const float* W1  = (const float*)d_in[3];  const float* b1  = (const float*)d_in[4];
  const float* W2  = (const float*)d_in[5];  const float* b2  = (const float*)d_in[6];
  const float* Wf1 = (const float*)d_in[7];  const float* bf1 = (const float*)d_in[8];
  const float* Wf2 = (const float*)d_in[9];  const float* bf2 = (const float*)d_in[10];
  const float* Wm1 = (const float*)d_in[11]; const float* bm1 = (const float*)d_in[12];
  const float* Wm2 = (const float*)d_in[13]; const float* bm2 = (const float*)d_in[14];

  const int* msrc = medges;
  const int* mdst = medges + ME_;

  float* ws     = (float*)d_ws;
  float* embs   = ws;                         // 1000*512
  float* pens   = embs + (size_t)G_ * 512;    // 1000
  float* dinv   = pens + G_;                  // 1000
  int*   deg    = (int*)(dinv + G_);          // 1000
  float* xw1    = (float*)(deg + G_);         // 1000*128
  float* hbuf   = xw1 + (size_t)G_ * MD_;     // 1000*128
  float* xwl    = hbuf + (size_t)G_ * MD_;    // 1000*10
  float* logits = xwl + (size_t)G_ * LB_;     // 1000*10
  float* out    = (float*)d_out;              // 10000 preds + 1 penalty

  // Stage 1: per-graph GNN (TDM ingest + WMMA GEMMs + LDS-atomic aggregation)
  seal_graph_kernel<<<G_, 256, 0, stream>>>(gx, gedges, W1, b1, W2, b2,
                                            Wf1, bf1, Wf2, bf2, embs, pens);

  // Stage 2: macro GCN
  seal_macro_deg_init<<<(G_ + 255) / 256, 256, 0, stream>>>(deg);
  seal_macro_deg_count<<<(ME_ + 255) / 256, 256, 0, stream>>>(mdst, deg);
  seal_macro_dinv<<<(G_ + 255) / 256, 256, 0, stream>>>(deg, dinv);

  seal_macro_gemm<<<(G_ + 63) / 64, 256, 0, stream>>>(embs, Wm1, xw1);
  seal_agg1_init<<<G_, MD_, 0, stream>>>(hbuf, xw1, dinv, bm1);
  seal_agg1_edges<<<ME_, MD_, 0, stream>>>(hbuf, xw1, dinv, msrc, mdst);

  seal_small_gemm<<<G_, MD_, 0, stream>>>(hbuf, Wm2, xwl);
  seal_agg2_init<<<G_, 32, 0, stream>>>(logits, xwl, dinv, bm2);
  seal_agg2_edges<<<ME_, 32, 0, stream>>>(logits, xwl, dinv, msrc, mdst);

  seal_logsoftmax<<<G_, 32, 0, stream>>>(logits, out);
  seal_pen_mean<<<1, 256, 0, stream>>>(pens, out + G_ * LB_);
}
